// Block_15650860827274
// MI455X (gfx1250) — matrix operand order repaired and checked by reading, not compiled, not run on hardware
//
#include <hip/hip_runtime.h>

typedef __attribute__((ext_vector_type(16))) _Float16 v16h;
typedef __attribute__((ext_vector_type(8)))  _Float16 v8h;
typedef __attribute__((ext_vector_type(8)))  float    v8f;
typedef __attribute__((ext_vector_type(4))) unsigned int u32x4;
typedef __attribute__((ext_vector_type(8))) int          i32x8;
typedef __attribute__((ext_vector_type(4))) int          i32x4;

#define B_  4
#define S_  512
#define D_  1024
#define H_  16
#define HD_ 64
#define F_  4096
#define M_TOK (B_ * S_)   // 2048

// ---------------------------------------------------------------------------
// WMMA helpers (CDNA5 wave32, 16x16x32 f16 -> f32)
// ---------------------------------------------------------------------------
__device__ inline v8f wmma16(v16h a, v16h b, v8f c) {
  return __builtin_amdgcn_wmma_f32_16x16x32_f16(
      false, a, false, b, (short)0, c, false, false);
}

// v16h fragment from two contiguous v8h (ds_load_b128) LDS loads.
// A-frag and B-frag: hiOff = 16 halves.
__device__ inline v16h ld_frag(const _Float16* p, int hiOff) {
  v8h lo = *(const v8h*)p;
  v8h hi = *(const v8h*)(p + hiOff);
  v16h r;
#pragma unroll
  for (int i = 0; i < 8; ++i) { r[i] = lo[i]; r[i + 8] = hi[i]; }
  return r;
}

__device__ inline float fast_rcp(float x) { return __builtin_amdgcn_rcpf(x); }

// ---------------------------------------------------------------------------
// Tensor Data Mover: 2D tile load (128 rows x 32 f16), LDS row stride 40
// halves via TDM padding (pad after every 16 dwords, pad 4 dwords).
// ---------------------------------------------------------------------------
__device__ inline i32x8 tdm_g1_32w(int K_elems) {
  // wg_mask=0 | data_size=1 (2B) | pad_enable | pad_interval=3 (16dw) | pad_amount=3 (4dw)
  unsigned d0 = (1u << 16) | (1u << 20) | (3u << 22) | (3u << 25);
  unsigned td0 = 0x100000u, td1 = 0x100000u;        // huge tensor dims (no OOB)
  unsigned long long s0 = (unsigned long long)K_elems;  // dim0 stride (elements)
  i32x8 g;
  g[0] = (int)d0;
  g[1] = (int)((td0 & 0xffffu) << 16);                           // abar=0 | tdim0.lo
  g[2] = (int)(((td0 >> 16) & 0xffffu) | ((td1 & 0xffffu) << 16));
  g[3] = (int)(((td1 >> 16) & 0xffffu) | (32u << 16));           // tile_dim0 = 32
  g[4] = (int)(128u);                                            // tile_dim1=128, tile_dim2=0
  g[5] = (int)(unsigned)(s0 & 0xffffffffu);
  g[6] = (int)(unsigned)((s0 >> 32) & 0xffffu);                  // stride1 = 0
  g[7] = 0;
  return g;
}

__device__ inline void tdm_load_tile(unsigned lds_byte_off, const _Float16* gsrc,
                                     i32x8 g1) {
  unsigned long long ga = (unsigned long long)(const void*)gsrc;
  u32x4 g0;
  g0.x = 1u;                                        // count=1, user mode
  g0.y = lds_byte_off;                              // lds_addr
  g0.z = (unsigned)(ga & 0xffffffffu);
  g0.w = (unsigned)((ga >> 32) & 0x1ffffffu) | (2u << 30);   // type=2 ("image")
  i32x4 z4 = {0, 0, 0, 0};
  i32x8 z8 = {0, 0, 0, 0, 0, 0, 0, 0};
  __builtin_amdgcn_tensor_load_to_lds(g0, g1, z4, z4, z8, 0);
}

// ---------------------------------------------------------------------------
// Weight convert + transpose: src f32 [K][N] (row-major) -> dst f16 [N][K]
// ---------------------------------------------------------------------------
__global__ __launch_bounds__(256) void transw_kernel(
    const float* __restrict__ src, _Float16* __restrict__ dst, int K, int N) {
  long long idx = (long long)blockIdx.x * 256 + threadIdx.x;
  if (idx >= (long long)K * N) return;
  int n = (int)(idx / K);
  int k = (int)(idx % K);
  dst[idx] = (_Float16)src[(size_t)k * N + n];
}

__global__ void bcat_kernel(const float* __restrict__ a, const float* __restrict__ b,
                            const float* __restrict__ c, float* __restrict__ o) {
  int i = blockIdx.x * 256 + threadIdx.x;
  if (i >= 3 * D_) return;
  o[i] = (i < D_) ? a[i] : (i < 2 * D_ ? b[i - D_] : c[i - 2 * D_]);
}

// ---------------------------------------------------------------------------
// Sinusoidal timestep embedding + SiLU
// ---------------------------------------------------------------------------
__global__ void emb_kernel(const int* __restrict__ ts, float* __restrict__ e) {
  int idx = blockIdx.x * 256 + threadIdx.x;
  if (idx >= B_ * D_) return;
  int b = idx >> 10, i = idx & 1023;
  float x = (float)ts[b] * 40.0f;                    // t / 100 * 4000
  int j = (i < 512) ? i : (i - 512);
  float fr = __expf(-9.210340371976184f * (float)j / 511.0f);
  float ang = x * fr;
  float v = (i < 512) ? __sinf(ang) : __cosf(ang);
  e[idx] = v * fast_rcp(1.0f + __expf(-v));          // silu
}

__global__ __launch_bounds__(256) void ada_kernel(
    const float* __restrict__ e, const float* __restrict__ W,
    const float* __restrict__ bv, float* __restrict__ out) {
  int idx = blockIdx.x * 256 + threadIdx.x;
  if (idx >= B_ * 2 * D_) return;
  int b = idx >> 11, j = idx & 2047;
  const float* ev = e + b * D_;
  float acc = bv[j];
  for (int k = 0; k < D_; ++k) acc += ev[k] * W[(size_t)k * 2048 + j];
  out[idx] = acc;
}

// ---------------------------------------------------------------------------
// LayerNorm. mode 0: AdaLN (p1 = ada[B][2048]: scale|shift). mode 1: p1=g, p2=b
// ---------------------------------------------------------------------------
__global__ __launch_bounds__(256) void ln_kernel(
    const float* __restrict__ in, const float* __restrict__ p1,
    const float* __restrict__ p2, float* __restrict__ outF,
    _Float16* __restrict__ outH, int mode) {
  __shared__ float rs[256], rq[256];
  int row = blockIdx.x;
  int b = row >> 9;
  const float* x = in + (size_t)row * D_;
  float s = 0.f, q = 0.f;
  for (int i = threadIdx.x; i < D_; i += 256) { float v = x[i]; s += v; q += v * v; }
  rs[threadIdx.x] = s; rq[threadIdx.x] = q;
  __syncthreads();
  for (int o = 128; o > 0; o >>= 1) {
    if (threadIdx.x < o) { rs[threadIdx.x] += rs[threadIdx.x + o]; rq[threadIdx.x] += rq[threadIdx.x + o]; }
    __syncthreads();
  }
  float mean = rs[0] * (1.0f / D_);
  float var  = rq[0] * (1.0f / D_) - mean * mean;
  float inv  = rsqrtf(var + 1e-5f);
  for (int i = threadIdx.x; i < D_; i += 256) {
    float v = (x[i] - mean) * inv;
    float y;
    if (mode == 0) y = v * (1.0f + p1[b * 2048 + i]) + p1[b * 2048 + D_ + i];
    else           y = v * p1[i] + p2[i];
    if (outF) outF[(size_t)row * D_ + i] = y;
    if (outH) outH[(size_t)row * D_ + i] = (_Float16)y;
  }
}

// ---------------------------------------------------------------------------
// Tiled WMMA GEMM: C[M,N] = A[M,K]f16 @ Bt[N,K]f16 (+bias [, gelu2] [, +res])
// Macro tile 128x128, 8 waves (2x4), 64x32 per wave, K-step 32.
// Tiles DMA'd by the Tensor Data Mover, double buffered on TENSORcnt.
// ---------------------------------------------------------------------------
#define LD_ST 40                    // padded LDS row stride (halves)
#define TILE_H (128 * LD_ST)        // one matrix tile in halves (5120)
#define BUF_H  (2 * TILE_H)         // A+B per buffer (10240 halves, 20480 B)

template <bool ACT, bool HASRES, bool OUTF, bool OUTH>
__global__ __launch_bounds__(256) void gemm_f16_kernel(
    const _Float16* __restrict__ A, const _Float16* __restrict__ Bt,
    const float* __restrict__ bias, const float* __restrict__ residual,
    float* __restrict__ outF, _Float16* __restrict__ outH,
    int M, int N, int K) {
  __shared__ _Float16 lds[2 * BUF_H];
  const int tid = threadIdx.x, lane = tid & 31, wave = tid >> 5;
  const int mw = wave >> 2, nw = wave & 3;
  const int mtile = blockIdx.y * 128, ntile = blockIdx.x * 128;
  const unsigned lds0 = (unsigned)(unsigned long long)(size_t)&lds[0];

  v8f acc[4][2] = {};
  const int akb = (lane & 16) ? 8 : 0;
  const int bkb = (lane & 16) ? 8 : 0;
  const int niter = K >> 5;

  const i32x8 g1 = tdm_g1_32w(K);
  const _Float16* Ab = A + (size_t)mtile * K;
  const _Float16* Bb = Bt + (size_t)ntile * K;

  if (wave == 0) {   // prologue: DMA tile 0 into buffer 0
    tdm_load_tile(lds0 + 0,              Ab + 0, g1);
    tdm_load_tile(lds0 + 2 * TILE_H,     Bb + 0, g1);
  }

  for (int i = 0; i < niter; ++i) {
    const int cur = i & 1;
    if (wave == 0) {
      if (i + 1 < niter) {   // prefetch next tile into other buffer
        unsigned bo = (cur ^ 1) * (unsigned)(2 * BUF_H);   // bytes
        tdm_load_tile(lds0 + bo,              Ab + (size_t)(i + 1) * 32, g1);
        tdm_load_tile(lds0 + bo + 2 * TILE_H, Bb + (size_t)(i + 1) * 32, g1);
        __builtin_amdgcn_s_wait_tensorcnt(2);   // oldest pair (cur buf) done
      } else {
        __builtin_amdgcn_s_wait_tensorcnt(0);
      }
    }
    __syncthreads();

    const _Float16* La = lds + cur * BUF_H;
    const _Float16* Lb = La + TILE_H;
    v16h af[4], bf[2];
#pragma unroll
    for (int f = 0; f < 4; ++f)
      af[f] = ld_frag(La + (mw * 64 + f * 16 + (lane & 15)) * LD_ST + akb, 16);
#pragma unroll
    for (int f = 0; f < 2; ++f)
      bf[f] = ld_frag(Lb + (nw * 32 + f * 16 + (lane & 15)) * LD_ST + bkb, 16);
#pragma unroll
    for (int fm = 0; fm < 4; ++fm)
#pragma unroll
      for (int fn = 0; fn < 2; ++fn)
        acc[fm][fn] = wmma16(af[fm], bf[fn], acc[fm][fn]);
    __syncthreads();
  }

  const int rb = (lane & 16) ? 8 : 0;
  const int nc = lane & 15;
#pragma unroll
  for (int fm = 0; fm < 4; ++fm)
#pragma unroll
    for (int fn = 0; fn < 2; ++fn)
#pragma unroll
      for (int r = 0; r < 8; ++r) {
        int gm = mtile + mw * 64 + fm * 16 + r + rb;
        int gn = ntile + nw * 32 + fn * 16 + nc;
        float v = acc[fm][fn][r] + bias[gn];
        if (ACT) v = v * fast_rcp(1.0f + __expf(-1.702f * v));   // gelu2
        if (HASRES) v += residual[(size_t)gm * N + gn];
        if (OUTF) outF[(size_t)gm * N + gn] = v;
        if (OUTH) outH[(size_t)gm * N + gn] = (_Float16)v;
      }
}

// ---------------------------------------------------------------------------
// Attention: one block per (b, h, 32-query-row tile). 256 threads, 8 waves.
// qkv layout: [token][3072] = q|k|v, head h at column h*64 within each third.
// ---------------------------------------------------------------------------
#define Q_ST  80
#define KT_ST 80
#define VT_ST 144
static constexpr int ATTN_SMEM =
    (32 * Q_ST) * 2 + 10240 * 2 + 32 * 512 * 4 + 32 * 512 * 2 +
    4 * 32 * 16 * 4 + 32 * 8 * 4;

__global__ __launch_bounds__(256) void attn_kernel(
    const _Float16* __restrict__ qkv, const float* __restrict__ abias,
    const unsigned char* __restrict__ mask, _Float16* __restrict__ ctx) {
  extern __shared__ char smem[];
  _Float16* ldsq  = (_Float16*)smem;            // [32][80]
  _Float16* ldskv = ldsq + 32 * Q_ST;           // kt [128][80] or vt [64][144]
  float*    s     = (float*)(ldskv + 10240);    // [32][512]
  _Float16* p     = (_Float16*)(s + 32 * 512);  // [32][512]
  float*    red   = (float*)(p + 32 * 512);     // [4][32][16]
  float*    rtmp  = red + 4 * 32 * 16;          // [32][8]

  const int tid  = threadIdx.x;
  const int lane = tid & 31;
  const int wave = tid >> 5;
  const int qt = blockIdx.x & 15;
  const int h  = (blockIdx.x >> 4) & 15;
  const int b  = blockIdx.x >> 8;
  const int tok0 = b * S_ + qt * 32;

  const int akb = (lane & 16) ? 8 : 0;
  const int bkb = (lane & 16) ? 8 : 0;
  const int rb  = (lane & 16) ? 8 : 0;
  const int nc  = lane & 15;

  for (int i = tid; i < 32 * 64; i += 256) {
    int r = i >> 6, d = i & 63;
    ldsq[r * Q_ST + d] = qkv[(size_t)(tok0 + r) * 3072 + h * HD_ + d];
  }

  // scores = (q/sqrt(d)) @ k^T, keys chunked 4 x 128
  for (int c = 0; c < 4; ++c) {
    for (int i = tid; i < 128 * 64; i += 256) {
      int kk = i >> 6, d = i & 63;
      ldskv[kk * KT_ST + d] =
          qkv[(size_t)(b * S_ + c * 128 + kk) * 3072 + D_ + h * HD_ + d];
    }
    __syncthreads();
    v8f sc0 = {}, sc1 = {};
#pragma unroll
    for (int ks = 0; ks < 64; ks += 32) {
      v16h a0 = ld_frag(ldsq + (0  + (lane & 15)) * Q_ST + ks + akb, 16);
      v16h a1 = ld_frag(ldsq + (16 + (lane & 15)) * Q_ST + ks + akb, 16);
      v16h bb = ld_frag(ldskv + (wave * 16 + (lane & 15)) * KT_ST + ks + bkb, 16);
      sc0 = wmma16(a0, bb, sc0);
      sc1 = wmma16(a1, bb, sc1);
    }
    int key = c * 128 + wave * 16 + nc;
#pragma unroll
    for (int r = 0; r < 8; ++r) {
      s[(r + rb) * 512 + key]      = sc0[r] * 0.125f;
      s[(16 + r + rb) * 512 + key] = sc1[r] * 0.125f;
    }
    __syncthreads();
  }

  // fp32 softmax with bias + mask (8 threads per row)
  {
    int row = tid >> 3, sub = tid & 7;
    int qg = qt * 32 + row;
    const float* bp = abias + (((size_t)(b * H_ + h) * S_ + qg) * S_);
    const unsigned char* mp = mask + ((size_t)b * S_ + qg) * S_;
    float mx = -3.0e38f;
    for (int j = 0; j < 64; ++j) {
      int col = sub * 64 + j;
      float v = s[row * 512 + col] + bp[col];
      if (mp[col]) v = -3.0e38f;
      s[row * 512 + col] = v;
      mx = fmaxf(mx, v);
    }
    rtmp[row * 8 + sub] = mx;
    __syncthreads();
    float m = rtmp[row * 8];
#pragma unroll
    for (int i = 1; i < 8; ++i) m = fmaxf(m, rtmp[row * 8 + i]);
    __syncthreads();
    float ssum = 0.f;
    for (int j = 0; j < 64; ++j) {
      int col = sub * 64 + j;
      float e2 = __expf(s[row * 512 + col] - m);
      s[row * 512 + col] = e2;
      ssum += e2;
    }
    rtmp[row * 8 + sub] = ssum;
    __syncthreads();
    float tot = 0.f;
#pragma unroll
    for (int i = 0; i < 8; ++i) tot += rtmp[row * 8 + i];
    float inv = fast_rcp(tot);
    for (int j = 0; j < 64; ++j) {
      int col = sub * 64 + j;
      p[row * 512 + col] = (_Float16)(s[row * 512 + col] * inv);
    }
    __syncthreads();
  }

  // ctx = probs @ v : wave w -> n-frag (w&3), K-half (w>>2); LDS split-K reduce
  v8f c0 = {}, c1 = {};
  const int jf = wave & 3, kh = wave >> 2;
  for (int c = 0; c < 4; ++c) {
    for (int i = tid; i < 64 * 128; i += 256) {          // vt transposed [d][key]
      int d = i >> 7, kk = i & 127;
      ldskv[d * VT_ST + kk] =
          qkv[(size_t)(b * S_ + c * 128 + kk) * 3072 + 2 * D_ + h * HD_ + d];
    }
    __syncthreads();
    if ((c >> 1) == kh) {
#pragma unroll
      for (int ks = 0; ks < 128; ks += 32) {
        int pkb = c * 128 + ks + akb;
        v16h a0 = ld_frag(p + (0  + (lane & 15)) * 512 + pkb, 16);
        v16h a1 = ld_frag(p + (16 + (lane & 15)) * 512 + pkb, 16);
        v16h bb = ld_frag(ldskv + (jf * 16 + (lane & 15)) * VT_ST + ks + bkb, 16);
        c0 = wmma16(a0, bb, c0);
        c1 = wmma16(a1, bb, c1);
      }
    }
    __syncthreads();
  }
  if (kh == 1) {
#pragma unroll
    for (int r = 0; r < 8; ++r) {
      red[(jf * 32 + r + rb) * 16 + nc]      = c0[r];
      red[(jf * 32 + 16 + r + rb) * 16 + nc] = c1[r];
    }
  }
  __syncthreads();
  if (kh == 0) {
#pragma unroll
    for (int r = 0; r < 8; ++r) {
      int m0 = r + rb, m1 = 16 + r + rb;
      float v0 = c0[r] + red[(jf * 32 + m0) * 16 + nc];
      float v1 = c1[r] + red[(jf * 32 + m1) * 16 + nc];
      ctx[(size_t)(tok0 + m0) * D_ + h * HD_ + jf * 16 + nc] = (_Float16)v0;
      ctx[(size_t)(tok0 + m1) * D_ + h * HD_ + jf * 16 + nc] = (_Float16)v1;
    }
  }
}

// ---------------------------------------------------------------------------
// Host launch
// ---------------------------------------------------------------------------
extern "C" void kernel_launch(void* const* d_in, const int* in_sizes, int n_in,
                              void* d_out, int out_size, void* d_ws, size_t ws_size,
                              hipStream_t stream) {
  const float* src   = (const float*)d_in[0];
  const unsigned char* smask = (const unsigned char*)d_in[1];
  const int*   tstep = (const int*)d_in[2];
  const float* abias = (const float*)d_in[3];
  const float* W_ada = (const float*)d_in[4];
  const float* b_ada = (const float*)d_in[5];
  const float* Wq = (const float*)d_in[6];  const float* bq = (const float*)d_in[7];
  const float* Wk = (const float*)d_in[8];  const float* bk = (const float*)d_in[9];
  const float* Wv = (const float*)d_in[10]; const float* bv = (const float*)d_in[11];
  const float* Wo = (const float*)d_in[12]; const float* bo = (const float*)d_in[13];
  const float* W1 = (const float*)d_in[14]; const float* b1 = (const float*)d_in[15];
  const float* W2 = (const float*)d_in[16]; const float* b2 = (const float*)d_in[17];
  const float* g2 = (const float*)d_in[18]; const float* be2 = (const float*)d_in[19];
  float* out = (float*)d_out;

  char* ws = (char*)d_ws;
  size_t off = 0;
  auto alloc = [&](size_t bytes) -> char* {
    char* pp = ws + off;
    off += (bytes + 255) & ~(size_t)255;
    return pp;
  };
  _Float16* wqkvT = (_Float16*)alloc((size_t)3 * D_ * D_ * 2);
  _Float16* woT   = (_Float16*)alloc((size_t)D_ * D_ * 2);
  _Float16* w1T   = (_Float16*)alloc((size_t)F_ * D_ * 2);
  _Float16* w2T   = (_Float16*)alloc((size_t)D_ * F_ * 2);
  float*    bqkv  = (float*)alloc(3 * D_ * 4);
  float*    ebuf  = (float*)alloc(B_ * D_ * 4);
  float*    ada   = (float*)alloc(B_ * 2 * D_ * 4);
  float*    x     = (float*)alloc((size_t)M_TOK * D_ * 4);
  _Float16* xh    = (_Float16*)alloc((size_t)M_TOK * D_ * 2);
  _Float16* qkvb  = (_Float16*)alloc((size_t)M_TOK * 3 * D_ * 2);
  _Float16* ctxb  = (_Float16*)alloc((size_t)M_TOK * D_ * 2);
  float*    x2    = (float*)alloc((size_t)M_TOK * D_ * 4);
  _Float16* hbuf  = (_Float16*)alloc((size_t)M_TOK * D_ * 2);
  _Float16* a1    = (_Float16*)alloc((size_t)M_TOK * F_ * 2);
  if (off > ws_size) return;

  const int T = 256;
  transw_kernel<<<(D_*D_)/T, T, 0, stream>>>(Wq, wqkvT,           D_, D_);
  transw_kernel<<<(D_*D_)/T, T, 0, stream>>>(Wk, wqkvT + D_*D_,   D_, D_);
  transw_kernel<<<(D_*D_)/T, T, 0, stream>>>(Wv, wqkvT + 2*D_*D_, D_, D_);
  transw_kernel<<<(D_*D_)/T, T, 0, stream>>>(Wo, woT,             D_, D_);
  transw_kernel<<<(D_*F_)/T, T, 0, stream>>>(W1, w1T,             D_, F_);
  transw_kernel<<<(F_*D_)/T, T, 0, stream>>>(W2, w2T,             F_, D_);
  bcat_kernel<<<(3*D_ + T-1)/T, T, 0, stream>>>(bq, bk, bv, bqkv);

  emb_kernel<<<(B_*D_ + T-1)/T, T, 0, stream>>>(tstep, ebuf);
  ada_kernel<<<(B_*2*D_)/T, T, 0, stream>>>(ebuf, W_ada, b_ada, ada);
  ln_kernel<<<M_TOK, T, 0, stream>>>(src, ada, nullptr, x, xh, 0);

  // QKV projection (fused N=3072): out f16 only
  gemm_f16_kernel<false, false, false, true><<<dim3(3*D_/128, M_TOK/128), T, 0, stream>>>(
      xh, wqkvT, bqkv, nullptr, nullptr, qkvb, M_TOK, 3*D_, D_);

  (void)hipFuncSetAttribute((const void*)attn_kernel,
                            hipFuncAttributeMaxDynamicSharedMemorySize, ATTN_SMEM);
  attn_kernel<<<B_ * H_ * (S_ / 32), T, ATTN_SMEM, stream>>>(qkvb, abias, smask, ctxb);

  // x2 = x + ctx @ Wo + bo : residual + f32 out
  gemm_f16_kernel<false, true, true, false><<<dim3(D_/128, M_TOK/128), T, 0, stream>>>(
      ctxb, woT, bo, x, x2, nullptr, M_TOK, D_, D_);

  ln_kernel<<<M_TOK, T, 0, stream>>>(x2, g2, be2, nullptr, hbuf, 1);

  // FFN1 with gelu2: f16 out
  gemm_f16_kernel<true, false, false, true><<<dim3(F_/128, M_TOK/128), T, 0, stream>>>(
      hbuf, w1T, b1, nullptr, nullptr, a1, M_TOK, F_, D_);

  // FFN2 + residual -> final f32 out
  gemm_f16_kernel<false, true, true, false><<<dim3(D_/128, M_TOK/128), T, 0, stream>>>(
      a1, w2T, b2, x2, out, nullptr, M_TOK, D_, F_);
}
